// StandardSplatAttention_31817117729237
// MI455X (gfx1250) — matrix operand order repaired and checked by reading, not compiled, not run on hardware
//
#include <hip/hip_runtime.h>
#include <hip/hip_bf16.h>

typedef __attribute__((ext_vector_type(16))) _Float16 v16h;
typedef __attribute__((ext_vector_type(8)))  _Float16 h8;
typedef __attribute__((ext_vector_type(4)))  _Float16 h4;
typedef __attribute__((ext_vector_type(8)))  float    v8f;
typedef __attribute__((ext_vector_type(4)))  int      v4i;

#define AS1 __attribute__((address_space(1)))
#define AS3 __attribute__((address_space(3)))

__device__ __forceinline__ v8f wmma16(v16h a, v16h b, v8f c) {
  return __builtin_amdgcn_wmma_f32_16x16x32_f16(
      false, a, false, b, (short)0, c, false, false);
}

__device__ __forceinline__ v16h cat16(h8 lo, h8 hi) {
  return __builtin_shufflevector(lo, hi, 0,1,2,3,4,5,6,7,8,9,10,11,12,13,14,15);
}

// ---- gfx1250 async global->LDS path (ASYNCcnt) with portable fallback ------
#if __has_builtin(__builtin_amdgcn_global_load_async_to_lds_b128)
#define HAVE_ASYNC_LDS 1
#else
#define HAVE_ASYNC_LDS 0
#endif

__device__ __forceinline__ void async_ld128(const void* g, void* lds) {
#if HAVE_ASYNC_LDS
  __builtin_amdgcn_global_load_async_to_lds_b128(
      (AS1 v4i*)g, (AS3 v4i*)lds, 0, 0);
#else
  *(h8*)lds = *(const h8*)g;
#endif
}

__device__ __forceinline__ void wait_async0() {
#if HAVE_ASYNC_LDS
#if __has_builtin(__builtin_amdgcn_s_wait_asynccnt)
  __builtin_amdgcn_s_wait_asynccnt(0);
#else
  asm volatile("s_wait_asynccnt 0x0" ::: "memory");
#endif
#endif
}

// ---------------------------------------------------------------------------
// C[M,N] = A[M,K] @ B[K,N] + bias[N], f32 in/out, f16 WMMA, double-buffered.
// 256 threads = 8 waves (4x2), BM=128 BN=64 BK=32, wave tile 32x32.
// ---------------------------------------------------------------------------
#define BM 128
#define BN 64
#define BK 32
#define KPAD 40   // 32+8 halfs: 80B row stride, keeps 16B alignment

__global__ __launch_bounds__(256) void gemm_wmma_f32(
    const float* __restrict__ A, const float* __restrict__ B,
    const float* __restrict__ bias, float* __restrict__ C,
    int M, int N, int K)
{
  __shared__ __attribute__((aligned(16))) _Float16 sA[2][BM * KPAD];   // [m][k]
  __shared__ __attribute__((aligned(16))) _Float16 sBt[2][BN * KPAD];  // [n][k]

  const int tid  = threadIdx.x;
  const int wave = tid >> 5, lane = tid & 31;
  const int lr = lane & 15, hi = lane >> 4, kb = hi << 3;
  const int m0 = blockIdx.y * BM, n0 = blockIdx.x * BN;
  const int wm = (wave & 3) * 32, wn = (wave >> 2) * 32;

  float4 ra[4], rb[2];

  auto loadA = [&](int k0) {
#pragma unroll
    for (int it = 0; it < 4; ++it) {
      int e = tid + it * 256;
      ra[it] = *(const float4*)(A + (size_t)(m0 + (e >> 3)) * K + k0 + ((e & 7) << 2));
    }
  };
  auto loadB = [&](int k0) {
#pragma unroll
    for (int it = 0; it < 2; ++it) {
      int e = tid + it * 256;
      rb[it] = *(const float4*)(B + (size_t)(k0 + (e >> 4)) * N + n0 + ((e & 15) << 2));
    }
  };
  auto stage = [&](int buf) {
#pragma unroll
    for (int it = 0; it < 4; ++it) {
      int e = tid + it * 256;
      h4 hv = { (_Float16)ra[it].x, (_Float16)ra[it].y,
                (_Float16)ra[it].z, (_Float16)ra[it].w };
      *(h4*)&sA[buf][(e >> 3) * KPAD + ((e & 7) << 2)] = hv;
    }
#pragma unroll
    for (int it = 0; it < 2; ++it) {
      int e = tid + it * 256;
      int r = e >> 4, c4 = (e & 15) << 2;
      sBt[buf][(c4 + 0) * KPAD + r] = (_Float16)rb[it].x;
      sBt[buf][(c4 + 1) * KPAD + r] = (_Float16)rb[it].y;
      sBt[buf][(c4 + 2) * KPAD + r] = (_Float16)rb[it].z;
      sBt[buf][(c4 + 3) * KPAD + r] = (_Float16)rb[it].w;
    }
  };

  v8f acc[2][2] = {};
  loadA(0); loadB(0); stage(0);
  __syncthreads();

  const int nk = K / BK;
  for (int ki = 0; ki < nk; ++ki) {
    const int cur = ki & 1;
    if (ki + 1 < nk) { loadA((ki + 1) * BK); loadB((ki + 1) * BK); }

    v16h af[2], bf[2];
#pragma unroll
    for (int i = 0; i < 2; ++i) {
      const _Float16* p = &sA[cur][(wm + i * 16 + lr) * KPAD];
      af[i] = cat16(*(const h8*)(p + hi * 16), *(const h8*)(p + hi * 16 + 8));
    }
#pragma unroll
    for (int j = 0; j < 2; ++j) {
      const _Float16* p = &sBt[cur][(wn + j * 16 + lr) * KPAD + hi * 16];
      bf[j] = cat16(*(const h8*)(p), *(const h8*)(p + 8));
    }
#pragma unroll
    for (int i = 0; i < 2; ++i)
#pragma unroll
      for (int j = 0; j < 2; ++j)
        acc[i][j] = wmma16(af[i], bf[j], acc[i][j]);

    if (ki + 1 < nk) stage(cur ^ 1);
    __syncthreads();
  }

#pragma unroll
  for (int i = 0; i < 2; ++i) {
    int mrow = m0 + wm + i * 16 + hi * 8;
#pragma unroll
    for (int j = 0; j < 2; ++j) {
      int ncol = n0 + wn + j * 16 + lr;
      float bv = bias ? bias[ncol] : 0.0f;
#pragma unroll
      for (int r = 0; r < 8; ++r)
        C[(size_t)(mrow + r) * N + ncol] = acc[i][j][r] + bv;
    }
  }
}

// ---------------------------------------------------------------------------
// Splat features on batch-mean q/k. qa[h][t][16] carries sigmoid(amp) factor,
// s=12..15 zero-padded. ka[h][t][16] without amplitude.
// ---------------------------------------------------------------------------
__global__ __launch_bounds__(256) void splat_feats(
    const float* __restrict__ qkv, const float* __restrict__ centers,
    const float* __restrict__ log_scales, const float* __restrict__ amps,
    _Float16* __restrict__ qa, _Float16* __restrict__ ka,
    int T, int H, int S)
{
  int gid = blockIdx.x * 256 + threadIdx.x;   // (h*T + t)*16 + s
  int s = gid & 15;
  int t = (gid >> 4) % T;
  int h = (gid >> 4) / T;
  if (h >= H) return;
  if (s >= S) { qa[gid] = (_Float16)0.0f; ka[gid] = (_Float16)0.0f; return; }

  size_t base0 = (size_t)t * 1536 + (size_t)h * 64;
  size_t base1 = ((size_t)T + t) * 1536 + (size_t)h * 64;
  const float* c = centers + (size_t)(h * S + s) * 64;
  float dq = 0.0f, dk = 0.0f;
#pragma unroll 8
  for (int d = 0; d < 64; ++d) {
    float qm = 0.5f * (qkv[base0 + d] + qkv[base1 + d]);
    float km = 0.5f * (qkv[base0 + 512 + d] + qkv[base1 + 512 + d]);
    float cc = c[d];
    float e1 = qm - cc; dq += e1 * e1;
    float e2 = km - cc; dk += e2 * e2;
  }
  float sc   = __expf(log_scales[h * S + s]) + 1e-8f;
  float inv2 = 1.0f / (sc * sc);
  float amp  = 1.0f / (1.0f + __expf(-amps[h * S + s]));
  qa[gid] = (_Float16)(amp * __expf(-0.5f * dq * inv2));
  ka[gid] = (_Float16)__expf(-0.5f * dk * inv2);
}

// ---------------------------------------------------------------------------
// Fused scores + softmax. Block = (h, 16-row t-tile); 8 waves x 16 j-tiles
// cover the full 2048-wide row in VGPRs (16 x v8f per lane). K padded 12->32.
// Row max/sum: shfl_xor within 16-lane groups, LDS across the 8 waves.
// Writes attn f32 (both identical batch slices) + probs f16 [h][t][j].
// ---------------------------------------------------------------------------
__global__ __launch_bounds__(256) void splat_scores_softmax(
    const _Float16* __restrict__ qa, const _Float16* __restrict__ ka,
    float* __restrict__ attn, _Float16* __restrict__ probs, int T, int H)
{
  const int tid  = threadIdx.x;
  const int wave = tid >> 5, lane = tid & 31;
  const int lr = lane & 15, hi = lane >> 4, kb = hi << 3;
  const int h  = blockIdx.y;
  const int ti = blockIdx.x;
  __shared__ float redbuf[8][16];

  h8 z8 = {};
  v16h z16 = {};
  const _Float16* qrow = qa + ((size_t)h * T + ti * 16 + lr) * 16;
  v16h a = cat16(*(const h8*)(qrow), *(const h8*)(qrow + 8));   // contiguous k; K 16..31 zeroed via b

  v8f acc[16];
#pragma unroll
  for (int tt = 0; tt < 16; ++tt) {
    int jt = wave + tt * 8;
    const _Float16* krow = ka + ((size_t)h * T + jt * 16 + lr) * 16;
    v16h b = cat16(*(const h8*)(krow), *(const h8*)(krow + 8));
    if (hi) b = z16;                             // lanes 16-31: K 16..31 -> 0
    v8f zc = {};
    acc[tt] = wmma16(a, b, zc);
  }

  // ---- row max ----
  float rm[8];
#pragma unroll
  for (int r = 0; r < 8; ++r) {
    float m = acc[0][r];
#pragma unroll
    for (int tt = 1; tt < 16; ++tt) m = fmaxf(m, acc[tt][r]);
    for (int msk = 1; msk < 16; msk <<= 1) m = fmaxf(m, __shfl_xor(m, msk));
    rm[r] = m;
  }
  if (lr == 0) {
#pragma unroll
    for (int r = 0; r < 8; ++r) redbuf[wave][hi * 8 + r] = rm[r];
  }
  __syncthreads();
#pragma unroll
  for (int r = 0; r < 8; ++r) {
    float m = redbuf[0][hi * 8 + r];
#pragma unroll
    for (int w = 1; w < 8; ++w) m = fmaxf(m, redbuf[w][hi * 8 + r]);
    rm[r] = m;
  }
  __syncthreads();

  // ---- exp + row sum ----
  float rs[8];
#pragma unroll
  for (int r = 0; r < 8; ++r) {
    float s = 0.0f;
#pragma unroll
    for (int tt = 0; tt < 16; ++tt) {
      float e = __expf(acc[tt][r] - rm[r]);
      acc[tt][r] = e;
      s += e;
    }
    for (int msk = 1; msk < 16; msk <<= 1) s += __shfl_xor(s, msk);
    rs[r] = s;
  }
  if (lr == 0) {
#pragma unroll
    for (int r = 0; r < 8; ++r) redbuf[wave][hi * 8 + r] = rs[r];
  }
  __syncthreads();
#pragma unroll
  for (int r = 0; r < 8; ++r) {
    float s = 0.0f;
#pragma unroll
    for (int w = 0; w < 8; ++w) s += redbuf[w][hi * 8 + r];
    rs[r] = 1.0f / s;
  }

  // ---- write attn (b=0 and b=1 identical) + f16 probs ----
  const size_t bstride = (size_t)T * T * H;
#pragma unroll
  for (int tt = 0; tt < 16; ++tt) {
    int j = (wave + tt * 8) * 16 + lr;
#pragma unroll
    for (int r = 0; r < 8; ++r) {
      int trow = ti * 16 + hi * 8 + r;
      float p = acc[tt][r] * rs[r];
      size_t idx = ((size_t)trow * T + j) * H + h;
      attn[idx] = p;
      attn[bstride + idx] = p;
      probs[((size_t)h * T + trow) * T + j] = (_Float16)p;
    }
  }
}

// ---------------------------------------------------------------------------
// outpre[b][t][h*64+d] = sum_j probs[h][t][j] * v[b][j][h][d]
// Probs tile (128x32 f16) staged via async global->LDS; V tile staged
// transposed via registers. Double-buffered.
// ---------------------------------------------------------------------------
__global__ __launch_bounds__(256) void attn_v_wmma(
    const _Float16* __restrict__ probs,   // [h][t][j]
    const float* __restrict__ qkv,        // [B*T][1536], v slice at +1024
    float* __restrict__ outpre, int T)
{
  const int b = blockIdx.z >> 3, h = blockIdx.z & 7;
  const int m0 = blockIdx.y * 128;

  __shared__ __attribute__((aligned(16))) _Float16 sAp[2][128 * KPAD];  // [t][k]
  __shared__ __attribute__((aligned(16))) _Float16 sVt[2][64 * KPAD];   // [d][k]

  const int tid  = threadIdx.x;
  const int wave = tid >> 5, lane = tid & 31;
  const int lr = lane & 15, hi = lane >> 4, kb = hi << 3;
  const int wm = (wave & 3) * 32, wn = (wave >> 2) * 32;

  float4 rv[2];
  auto asyncA = [&](int k0, int buf) {
#pragma unroll
    for (int it = 0; it < 2; ++it) {
      int c = tid + it * 256;          // 512 x 16B chunks: 128 rows x 4
      int row = c >> 2, part = c & 3;
      const _Float16* g = probs + ((size_t)h * T + m0 + row) * T + k0 + part * 8;
      async_ld128(g, &sAp[buf][row * KPAD + part * 8]);
    }
  };
  auto loadV = [&](int k0) {
#pragma unroll
    for (int it = 0; it < 2; ++it) {
      int e = tid + it * 256;
      rv[it] = *(const float4*)(qkv + ((size_t)(b * T + k0 + (e >> 4)) * 3 + 2) * 512
                                    + h * 64 + ((e & 15) << 2));
    }
  };
  auto stageV = [&](int buf) {
#pragma unroll
    for (int it = 0; it < 2; ++it) {
      int e = tid + it * 256;
      int j = e >> 4, dq = (e & 15) << 2;
      sVt[buf][(dq + 0) * KPAD + j] = (_Float16)rv[it].x;
      sVt[buf][(dq + 1) * KPAD + j] = (_Float16)rv[it].y;
      sVt[buf][(dq + 2) * KPAD + j] = (_Float16)rv[it].z;
      sVt[buf][(dq + 3) * KPAD + j] = (_Float16)rv[it].w;
    }
  };

  v8f acc[2][2] = {};
  asyncA(0, 0); loadV(0); stageV(0);
  wait_async0();
  __syncthreads();

  const int nk = T / 32;
  for (int ki = 0; ki < nk; ++ki) {
    const int cur = ki & 1;
    if (ki + 1 < nk) { asyncA((ki + 1) * 32, cur ^ 1); loadV((ki + 1) * 32); }

    v16h af[2], bf[2];
#pragma unroll
    for (int i = 0; i < 2; ++i) {
      const _Float16* p = &sAp[cur][(wm + i * 16 + lr) * KPAD];
      af[i] = cat16(*(const h8*)(p + hi * 16), *(const h8*)(p + hi * 16 + 8));
    }
#pragma unroll
    for (int j = 0; j < 2; ++j) {
      const _Float16* p = &sVt[cur][(wn + j * 16 + lr) * KPAD + hi * 16];
      bf[j] = cat16(*(const h8*)(p), *(const h8*)(p + 8));
    }
#pragma unroll
    for (int i = 0; i < 2; ++i)
#pragma unroll
      for (int j = 0; j < 2; ++j)
        acc[i][j] = wmma16(af[i], bf[j], acc[i][j]);

    if (ki + 1 < nk) stageV(cur ^ 1);
    wait_async0();
    __syncthreads();
  }

#pragma unroll
  for (int i = 0; i < 2; ++i) {
    int trow = m0 + wm + i * 16 + hi * 8;
#pragma unroll
    for (int j = 0; j < 2; ++j) {
      int n = wn + j * 16 + lr;
#pragma unroll
      for (int r = 0; r < 8; ++r)
        outpre[((size_t)(b * T + trow + r)) * 512 + h * 64 + n] = acc[i][j][r];
    }
  }
}

// ---------------------------------------------------------------------------
extern "C" void kernel_launch(void* const* d_in, const int* in_sizes, int n_in,
                              void* d_out, int out_size, void* d_ws, size_t ws_size,
                              hipStream_t stream) {
  (void)in_sizes; (void)n_in; (void)out_size; (void)ws_size;

  const float* x       = (const float*)d_in[0];
  const float* centers = (const float*)d_in[1];
  const float* lsc     = (const float*)d_in[2];
  const float* amp     = (const float*)d_in[3];
  const float* Wqkv    = (const float*)d_in[4];
  const float* bqkv    = (const float*)d_in[5];
  const float* Wout    = (const float*)d_in[6];
  const float* bout    = (const float*)d_in[7];
  float* out = (float*)d_out;

  const int B = 2, T = 2048, D = 512, H = 8, S = 12;
  const int M = B * T;  // 4096

  char* ws = (char*)d_ws;
  float*    qkv    = (float*)ws;                           // 24 MiB
  size_t off = (size_t)M * 1536 * sizeof(float);
  _Float16* qa     = (_Float16*)(ws + off); off += (size_t)H * T * 16 * 2;
  _Float16* ka     = (_Float16*)(ws + off); off += (size_t)H * T * 16 * 2;
  _Float16* probs  = (_Float16*)(ws + off); off += (size_t)H * T * T * 2;  // 64 MiB
  float*    outpre = (float*)(ws + off);                   // 8 MiB

  float* attn = out + (size_t)B * T * D;

  gemm_wmma_f32<<<dim3(1536 / BN, M / BM), 256, 0, stream>>>(
      x, Wqkv, bqkv, qkv, M, 1536, 512);

  splat_feats<<<(H * T * 16) / 256, 256, 0, stream>>>(
      qkv, centers, lsc, amp, qa, ka, T, H, S);

  splat_scores_softmax<<<dim3(T / 16, H), 256, 0, stream>>>(
      qa, ka, attn, probs, T, H);

  attn_v_wmma<<<dim3(1, T / 128, B * H), 256, 0, stream>>>(probs, qkv, outpre, T);

  gemm_wmma_f32<<<dim3(512 / BN, M / BM), 256, 0, stream>>>(
      outpre, Wout, bout, out, M, 512, 512);
}
